// MultiHeadedAttention_86492051406926
// MI455X (gfx1250) — matrix operand order repaired and checked by reading, not compiled, not run on hardware
//
#include <hip/hip_runtime.h>
#include <hip/hip_bf16.h>
#include <stdint.h>

// ---------------------------------------------------------------------------
// Sizes (fixed by the reference)
// ---------------------------------------------------------------------------
#define BB 8
#define NN 1024
#define DD 512
#define HH 8
#define DK 64
#define BN (BB * NN)       // 8192 tokens
#define ALPHA 0.2f
#define PPAD 8             // bf16 row padding for LDS p-tile (bank spread)

typedef __attribute__((ext_vector_type(16))) __bf16 v16bf;
typedef __attribute__((ext_vector_type(8)))  __bf16 v8bf;
typedef __attribute__((ext_vector_type(8)))  float  v8f;

// D = A(16x32 bf16) x B(32x16 bf16) + C(16x16 f32)   (wave32, whole-wave op)
__device__ __forceinline__ v8f wmma_bf16(v16bf a, v16bf b, v8f c) {
  return __builtin_amdgcn_wmma_f32_16x16x32_bf16(
      /*neg_a=*/false, a, /*neg_b=*/false, b,
      /*c_mod=*/(short)0, c, /*reuse_a=*/false, /*reuse_b=*/false);
}

// A-fragment (16x32, K-major rows): lane L holds row M=L&15;
// lanes 0-15 K in {0..15}, lanes 16-31 K in {16..31};
// element e -> K = 16*(L>>4) + e (matches B-fragment order).
__device__ __forceinline__ v16bf load_a_frag(const __bf16* base, int ld,
                                             int kbase, int lane) {
  int r = lane & 15, kh = lane >> 4;
  const __bf16* p = base + r * ld + kbase + 16 * kh;
  v8bf lo = *(const v8bf*)(p);
  v8bf hi = *(const v8bf*)(p + 8);
  return __builtin_shufflevector(lo, hi, 0, 1, 2, 3, 4, 5, 6, 7, 8, 9, 10, 11,
                                 12, 13, 14, 15);
}

// B-fragment (32x16, stored K-major per output column n): lane L holds
// column N=L&15, element e -> K = 16*(L>>4) + e.  One aligned 32B load.
__device__ __forceinline__ v16bf load_b_frag(const __bf16* base, int ld,
                                             int kbase, int lane) {
  int n = lane & 15, kh = lane >> 4;
  return *(const v16bf*)(base + n * ld + kbase + 16 * kh);
}

// Async global->LDS copy of one DWORD (GVS mode), tracked by ASYNCcnt.
__device__ __forceinline__ void async_ld_b32(uint32_t lds_off, uint32_t goff,
                                             const void* sbase) {
  asm volatile("global_load_async_to_lds_b32 %0, %1, %2"
               :: "v"(lds_off), "v"(goff), "s"(sbase)
               : "memory");
}
__device__ __forceinline__ void wait_asynccnt0() {
  asm volatile("s_wait_asynccnt 0" ::: "memory");
}

// ---------------------------------------------------------------------------
// 1) fp32 -> bf16 conversion (value, Wv, Wo)
// ---------------------------------------------------------------------------
__global__ void cvt_bf16_kernel(const float* __restrict__ src,
                                __bf16* __restrict__ dst, int n) {
  int i = blockIdx.x * 256 + threadIdx.x;
  if (i < n) dst[i] = (__bf16)src[i];
}

// ---------------------------------------------------------------------------
// 2) Fold the attention vector into the Q/K projections:
//    aq[h,k] = sum_dk Wq[h*64+dk, k] * a[h, dk]     (ditto ak / bqa / bka)
// ---------------------------------------------------------------------------
__global__ void fold_kernel(const float* __restrict__ Wq,
                            const float* __restrict__ Wk,
                            const float* __restrict__ bq,
                            const float* __restrict__ bk,
                            const float* __restrict__ a,
                            float* __restrict__ aq, float* __restrict__ ak,
                            float* __restrict__ bqa, float* __restrict__ bka) {
  int idx = blockIdx.x * 256 + threadIdx.x;
  if (idx >= HH * DD) return;
  int h = idx >> 9, k = idx & (DD - 1);
  float s1 = 0.f, s2 = 0.f;
  for (int dk = 0; dk < DK; ++dk) {
    s1 += Wq[(h * DK + dk) * DD + k] * a[h * 2 * DK + dk];
    s2 += Wk[(h * DK + dk) * DD + k] * a[h * 2 * DK + DK + dk];
  }
  aq[idx] = s1;
  ak[idx] = s2;
  if (k == 0) {
    float t1 = 0.f, t2 = 0.f;
    for (int dk = 0; dk < DK; ++dk) {
      t1 += bq[h * DK + dk] * a[h * 2 * DK + dk];
      t2 += bk[h * DK + dk] * a[h * 2 * DK + DK + dk];
    }
    bqa[h] = t1;
    bka[h] = t2;
  }
}

// ---------------------------------------------------------------------------
// 3) sq[b,h,n] = query[b,n,:] . aq[h,:] + bqa[h]   (one wave32 per token row,
//    8 heads per lane, cross-lane reduce with __shfl_xor)
// ---------------------------------------------------------------------------
__global__ __launch_bounds__(256) void sqsk_kernel(
    const float* __restrict__ query, const float* __restrict__ key,
    const float* __restrict__ aq, const float* __restrict__ ak,
    const float* __restrict__ bqa, const float* __restrict__ bka,
    float* __restrict__ sq, float* __restrict__ sk) {
  __shared__ float saq[HH * DD];   // 16 KB
  __shared__ float sak[HH * DD];   // 16 KB
  for (int i = threadIdx.x; i < HH * DD; i += 256) {
    saq[i] = aq[i];
    sak[i] = ak[i];
  }
  __syncthreads();
  int lane = threadIdx.x & 31, w = threadIdx.x >> 5;
  int m = blockIdx.x * 8 + w;                 // token row (0..8191)
  const float* qrow = query + (size_t)m * DD;
  const float* krow = key + (size_t)m * DD;
  float accq[HH] = {}, acck[HH] = {};
  for (int k = lane; k < DD; k += 32) {
    float qv = qrow[k], kv = krow[k];
#pragma unroll
    for (int h = 0; h < HH; ++h) {
      accq[h] += qv * saq[h * DD + k];
      acck[h] += kv * sak[h * DD + k];
    }
  }
#pragma unroll
  for (int h = 0; h < HH; ++h)
    for (int off = 16; off > 0; off >>= 1) {
      accq[h] += __shfl_xor(accq[h], off, 32);
      acck[h] += __shfl_xor(acck[h], off, 32);
    }
  if (lane == 0) {
    int b = m >> 10, n = m & (NN - 1);
#pragma unroll
    for (int h = 0; h < HH; ++h) {
      sq[(b * HH + h) * NN + n] = accq[h] + bqa[h];
      sk[(b * HH + h) * NN + n] = acck[h] + bka[h];
    }
  }
}

// ---------------------------------------------------------------------------
// 4) V projection with WMMA, writing TRANSPOSED bf16:
//    vt[(b*H+h)*DK+d][j] = value[b,j,:].Wv[h*64+d,:] + bv   (K-major for p@v)
// ---------------------------------------------------------------------------
__global__ __launch_bounds__(128) void vproj_kernel(
    const __bf16* __restrict__ Abf, const __bf16* __restrict__ Wbf,
    const float* __restrict__ bias, __bf16* __restrict__ vt) {
  int lane = threadIdx.x & 31, w = threadIdx.x >> 5;
  int mbase = blockIdx.x * 16;
  int nbase = blockIdx.y * 64 + w * 16;
  const __bf16* arow = Abf + (size_t)mbase * DD;
  const __bf16* brow = Wbf + (size_t)nbase * DD;
  v8f acc = {};
  for (int kb = 0; kb < DD; kb += 32) {
    if (kb + 32 < DD) {  // -> global_prefetch_b8
      __builtin_prefetch(arow + (lane & 15) * DD + kb + 32, 0, 1);
      __builtin_prefetch(brow + (lane & 15) * DD + kb + 32, 0, 1);
    }
    acc = wmma_bf16(load_a_frag(arow, DD, kb, lane),
                    load_b_frag(brow, DD, kb, lane), acc);
  }
  int n = nbase + (lane & 15);
  float bv = bias[n];
  int h = n >> 6, d = n & (DK - 1);
  int kh = lane >> 4;
#pragma unroll
  for (int r = 0; r < 8; ++r) {  // C/D: M = r + 8*(lane>>4), N = lane&15
    int m = mbase + r + 8 * kh;
    int b = m >> 10, j = m & (NN - 1);
    vt[(size_t)((b * HH + h) * DK + d) * NN + j] = (__bf16)(acc[r] + bv);
  }
}

// ---------------------------------------------------------------------------
// 5) Attention: 16 query rows x one (b,h) per workgroup; 4 waves.
//    Pass1: per-row max & sum over unmasked j.
//    Pass2a: 128 threads build the bf16 p-tile ONCE in LDS (no 4x redundant
//            exp), rows padded so A-fragment ds_load_b128s are conflict-free.
//    Pass2b: wave w computes d-tile w: acc += WMMA(p[16x32], vT[32x16]).
// ---------------------------------------------------------------------------
__global__ __launch_bounds__(128) void attn_kernel(
    const float* __restrict__ sqg, const float* __restrict__ skg,
    const int* __restrict__ mask, const __bf16* __restrict__ vt,
    __bf16* __restrict__ xbf) {
  __shared__ float sks[NN];                 // 4 KB  (async-filled)
  __shared__ unsigned char mk[16][NN];      // 16 KB
  __shared__ __bf16 pbf[16][NN + PPAD];     // ~32 KB, padded rows
  __shared__ float ms[16], is[16], sqs[16];
  int tid = threadIdx.x, lane = tid & 31, w = tid >> 5;
  int bh = blockIdx.y, b = bh >> 3, h = bh & 7;
  int ibase = blockIdx.x * 16;

  // ---- async global->LDS stage of sk row (ASYNCcnt path) ----
  {
    uint32_t lds0 = (uint32_t)(uintptr_t)(&sks[0]);
    uint32_t gbase = (uint32_t)(bh * NN) * 4u;
    for (int j = tid; j < NN; j += 128)
      async_ld_b32(lds0 + 4u * (uint32_t)j, gbase + 4u * (uint32_t)j, skg);
  }
  if (tid < 16) sqs[tid] = sqg[bh * NN + ibase + tid];
  const int* mrow = mask + ((size_t)(b * NN + ibase)) * NN;
  for (int idx = tid; idx < 16 * NN; idx += 128)
    mk[idx >> 10][idx & (NN - 1)] = (unsigned char)(mrow[idx] != 0);
  wait_asynccnt0();
  __syncthreads();

  // ---- pass 1: row statistics (wave w handles rows 4w..4w+3) ----
  for (int q = 0; q < 4; ++q) {
    int r = w * 4 + q;
    float sqi = sqs[r];
    float mx = -1e30f;
    for (int j = lane; j < NN; j += 32)
      if (mk[r][j]) {
        float s = sqi + sks[j];
        s = s >= 0.f ? s : ALPHA * s;
        mx = fmaxf(mx, s);
      }
    for (int off = 16; off > 0; off >>= 1) mx = fmaxf(mx, __shfl_xor(mx, off, 32));
    float sum = 0.f;
    for (int j = lane; j < NN; j += 32)
      if (mk[r][j]) {
        float s = sqi + sks[j];
        s = s >= 0.f ? s : ALPHA * s;
        sum += __expf(s - mx);
      }
    for (int off = 16; off > 0; off >>= 1) sum += __shfl_xor(sum, off, 32);
    if (lane == 0) {
      ms[r] = mx;
      is[r] = sum > 0.f ? 1.f / sum : 0.f;
    }
  }
  __syncthreads();

  // ---- pass 2a: build the 16x1024 bf16 probability tile once ----
  for (int idx = tid; idx < 16 * NN; idx += 128) {
    int r = idx >> 10, j = idx & (NN - 1);
    float pv = 0.f;
    if (mk[r][j]) {
      float s = sqs[r] + sks[j];
      s = s >= 0.f ? s : ALPHA * s;
      pv = __expf(s - ms[r]) * is[r];   // fully-masked rows -> pv stays 0
    }
    pbf[r][j] = (__bf16)pv;
  }
  __syncthreads();

  // ---- pass 2b: p @ v^T with WMMA (A from LDS, B streamed from L2) ----
  int rA = lane & 15, kh = lane >> 4;
  int d = w * 16 + rA;  // B-frag column = lane&15
  const __bf16* vrow = vt + (size_t)(bh * DK + d) * NN;
  v8f acc = {};
  for (int jb = 0; jb < NN; jb += 32) {
    if (jb + 32 < NN) __builtin_prefetch(vrow + jb + 32 + kh * 16, 0, 1);
    // A-fragment from LDS: row rA, K runs {jb+16kh..+15}
    const __bf16* pp = &pbf[rA][jb + 16 * kh];
    v8bf lo = *(const v8bf*)(pp);
    v8bf hi = *(const v8bf*)(pp + 8);
    v16bf pa = __builtin_shufflevector(lo, hi, 0, 1, 2, 3, 4, 5, 6, 7, 8, 9,
                                       10, 11, 12, 13, 14, 15);
    v16bf bb = *(const v16bf*)(vrow + jb + kh * 16);
    acc = wmma_bf16(pa, bb, acc);
  }
#pragma unroll
  for (int r = 0; r < 8; ++r) {
    int i = ibase + r + 8 * kh;
    // x layout [b][n][h][dk] == row-major [token][feature] for the out GEMM
    xbf[(size_t)((b * NN + i) * HH + h) * DK + d] = (__bf16)acc[r];
  }
}

// ---------------------------------------------------------------------------
// 6) Output projection: out[m,n] = x[m,:].Wo[n,:] + bo[n]  (fp32 result)
// ---------------------------------------------------------------------------
__global__ __launch_bounds__(128) void outproj_kernel(
    const __bf16* __restrict__ Xbf, const __bf16* __restrict__ Wobf,
    const float* __restrict__ bo, float* __restrict__ out) {
  int lane = threadIdx.x & 31, w = threadIdx.x >> 5;
  int mbase = blockIdx.x * 16;
  int nbase = blockIdx.y * 64 + w * 16;
  const __bf16* arow = Xbf + (size_t)mbase * DD;
  const __bf16* brow = Wobf + (size_t)nbase * DD;
  v8f acc = {};
  for (int kb = 0; kb < DD; kb += 32) {
    if (kb + 32 < DD) {
      __builtin_prefetch(arow + (lane & 15) * DD + kb + 32, 0, 1);
      __builtin_prefetch(brow + (lane & 15) * DD + kb + 32, 0, 1);
    }
    acc = wmma_bf16(load_a_frag(arow, DD, kb, lane),
                    load_b_frag(brow, DD, kb, lane), acc);
  }
  int n = nbase + (lane & 15);
  float bias = bo[n];
  int kh = lane >> 4;
#pragma unroll
  for (int r = 0; r < 8; ++r) {
    int m = mbase + r + 8 * kh;
    out[(size_t)m * DD + n] = acc[r] + bias;
  }
}

// ---------------------------------------------------------------------------
// Host launcher
// ---------------------------------------------------------------------------
extern "C" void kernel_launch(void* const* d_in, const int* in_sizes, int n_in,
                              void* d_out, int out_size, void* d_ws,
                              size_t ws_size, hipStream_t stream) {
  const float* query = (const float*)d_in[0];
  const float* key   = (const float*)d_in[1];
  const float* value = (const float*)d_in[2];
  const int*   mask  = (const int*)d_in[3];
  const float* Wq = (const float*)d_in[4];
  const float* bq = (const float*)d_in[5];
  const float* Wk = (const float*)d_in[6];
  const float* bk = (const float*)d_in[7];
  const float* Wv = (const float*)d_in[8];
  const float* bv = (const float*)d_in[9];
  const float* Wo = (const float*)d_in[10];
  const float* bo = (const float*)d_in[11];
  const float* a  = (const float*)d_in[12];
  float* out = (float*)d_out;

  // workspace carve-up (256B aligned sections)
  char* base = (char*)d_ws;
  size_t off = 0;
  auto alloc = [&](size_t bytes) {
    char* p = base + off;
    off += (bytes + 255) & ~(size_t)255;
    return p;
  };
  __bf16* value_bf = (__bf16*)alloc((size_t)BN * DD * 2);   // 8 MB
  __bf16* Wv_bf    = (__bf16*)alloc((size_t)DD * DD * 2);   // 512 KB
  __bf16* Wo_bf    = (__bf16*)alloc((size_t)DD * DD * 2);   // 512 KB
  float*  aq  = (float*)alloc((size_t)HH * DD * 4);
  float*  ak  = (float*)alloc((size_t)HH * DD * 4);
  float*  bqa = (float*)alloc(HH * 4);
  float*  bka = (float*)alloc(HH * 4);
  float*  sq  = (float*)alloc((size_t)BB * HH * NN * 4);    // 256 KB
  float*  sk  = (float*)alloc((size_t)BB * HH * NN * 4);    // 256 KB
  __bf16* vt  = (__bf16*)alloc((size_t)BB * HH * DK * NN * 2); // 8 MB
  __bf16* xbf = (__bf16*)alloc((size_t)BN * DD * 2);        // 8 MB
  (void)ws_size; (void)n_in; (void)in_sizes; (void)out_size;

  // 1) bf16 conversions
  cvt_bf16_kernel<<<(BN * DD + 255) / 256, 256, 0, stream>>>(value, value_bf,
                                                             BN * DD);
  cvt_bf16_kernel<<<(DD * DD + 255) / 256, 256, 0, stream>>>(Wv, Wv_bf,
                                                             DD * DD);
  cvt_bf16_kernel<<<(DD * DD + 255) / 256, 256, 0, stream>>>(Wo, Wo_bf,
                                                             DD * DD);
  // 2) fold attention vector into Q/K weights
  fold_kernel<<<(HH * DD + 255) / 256, 256, 0, stream>>>(Wq, Wk, bq, bk, a, aq,
                                                         ak, bqa, bka);
  // 3) sq / sk GEMVs
  sqsk_kernel<<<BN / 8, 256, 0, stream>>>(query, key, aq, ak, bqa, bka, sq, sk);
  // 4) V projection (WMMA) -> transposed bf16
  vproj_kernel<<<dim3(BN / 16, DD / 64), 128, 0, stream>>>(value_bf, Wv_bf, bv,
                                                           vt);
  // 5) attention + p@v (WMMA)
  attn_kernel<<<dim3(NN / 16, BB * HH), 128, 0, stream>>>(sq, sk, mask, vt,
                                                          xbf);
  // 6) output projection (WMMA) -> fp32
  outproj_kernel<<<dim3(BN / 16, DD / 64), 128, 0, stream>>>(xbf, Wo_bf, bo,
                                                             out);
}
